// ScalarDotProductCriticNetworkDualAttention_18614388261537
// MI455X (gfx1250) — matrix operand order repaired and checked by reading, not compiled, not run on hardware
//
#include <hip/hip_runtime.h>
#include <hip/hip_bf16.h>

typedef __attribute__((ext_vector_type(16))) _Float16 v16h;
typedef __attribute__((ext_vector_type(8)))  float    v8f;

union ABfrag { v16h v; _Float16 h[16]; uint4 q[2]; };

// ---------------------------------------------------------------------------
// Fast transcendentals: native v_tanh_f32 on gfx1250 if the builtin exists,
// else branch-free exp-based fallback (no libm EXEC-mask gymnastics).
// ---------------------------------------------------------------------------
__device__ inline float fast_tanh(float x) {
#if __has_builtin(__builtin_amdgcn_tanhf)
  return __builtin_amdgcn_tanhf(x);
#elif __has_builtin(__builtin_amdgcn_tanh_f32)
  return __builtin_amdgcn_tanh_f32(x);
#else
  float e = __expf(2.0f * x);
  return 1.0f - __fdividef(2.0f, e + 1.0f);
#endif
}

// ---------------------------------------------------------------------------
// Fragment loaders (wave32, 16x16x32 f16 WMMA)
// A (16x32 MxK) [ISA 7.12.2]: lane L: M=L%16, hsel=L/16;
//   elems 0..15 -> K=k0+16h+(0..15) (contiguous, same order as B)
// B (32x16 KxN): lane L: N=L%16; elem e -> K = k0 + 16*(L/16) + e
// C/D (16x16 f32): lane L, vgpr r -> M = r + 8*(L/16), N = L%16
// ---------------------------------------------------------------------------
__device__ inline v16h load_A_lds(const _Float16* A, int ld, int k0) {
  int lane = threadIdx.x & 31;
  int m = lane & 15, h = lane >> 4;
  const _Float16* p = A + m * ld + k0 + 16 * h;
  ABfrag a;
  a.q[0] = *(const uint4*)(p);
  a.q[1] = *(const uint4*)(p + 8);
  return a.v;
}

__device__ inline v16h load_B_ldsT(const _Float16* rows, int ld, int k0, int n0) {
  int lane = threadIdx.x & 31;
  int n = lane & 15, h = lane >> 4;
  const _Float16* p = rows + (n0 + n) * ld + k0 + 16 * h;
  ABfrag b;
  b.q[0] = *(const uint4*)(p);
  b.q[1] = *(const uint4*)(p + 8);
  return b.v;
}

__device__ inline v16h load_B_ldscol(const _Float16* mat, int ld, int k0, int n0, int krows) {
  int lane = threadIdx.x & 31;
  int n = n0 + (lane & 15);
  int kb = k0 + 16 * (lane >> 4);
  ABfrag b;
#pragma unroll
  for (int e = 0; e < 16; ++e) {
    int kk = kb + e;
    b.h[e] = (kk < krows) ? mat[kk * ld + n] : (_Float16)0.0f;
  }
  return b.v;
}

__device__ inline v16h load_B_frag(const _Float16* frag) {
  int lane = threadIdx.x & 31;
  const _Float16* p = frag + lane * 16;
  ABfrag b;
  b.q[0] = *(const uint4*)(p);
  b.q[1] = *(const uint4*)(p + 8);
  return b.v;
}

template <bool TANH>
__device__ inline void store_D_f16(v8f c, _Float16* D, int ld, int n0) {
  int lane = threadIdx.x & 31;
  int n = lane & 15, h = lane >> 4;
#pragma unroll
  for (int r = 0; r < 8; ++r) {
    float v = c[r];
    if (TANH) v = fast_tanh(v);
    D[(r + 8 * h) * ld + n0 + n] = (_Float16)v;
  }
}

__device__ inline void store_D_f32(v8f c, float* D, int ld, int n0) {
  int lane = threadIdx.x & 31;
  int n = lane & 15, h = lane >> 4;
#pragma unroll
  for (int r = 0; r < 8; ++r) D[(r + 8 * h) * ld + n0 + n] = c[r];
}

__device__ inline v8f wmma16(v16h a, v16h b, v8f c) {
  return __builtin_amdgcn_wmma_f32_16x16x32_f16(false, a, false, b, (short)0, c, false, false);
}

// Unrolled K=128 GEMM vs pre-swizzled global fragments: all loads first, then WMMA chain.
__device__ inline v8f gemm4_frag(const _Float16* A, int lda, const _Float16* fb, long step, v8f c) {
  v16h a0 = load_A_lds(A, lda, 0);
  v16h a1 = load_A_lds(A, lda, 32);
  v16h a2 = load_A_lds(A, lda, 64);
  v16h a3 = load_A_lds(A, lda, 96);
  v16h b0 = load_B_frag(fb);
  v16h b1 = load_B_frag(fb + step);
  v16h b2 = load_B_frag(fb + 2 * step);
  v16h b3 = load_B_frag(fb + 3 * step);
  c = wmma16(a0, b0, c);
  c = wmma16(a1, b1, c);
  c = wmma16(a2, b2, c);
  c = wmma16(a3, b3, c);
  return c;
}

// K=160 (128 + 32-wide zero-padded tail from side array Ap)
__device__ inline v8f gemm5_frag(const _Float16* A, int lda, const _Float16* Ap,
                                 const _Float16* fb, long step, v8f c) {
  v16h a0 = load_A_lds(A, lda, 0);
  v16h a1 = load_A_lds(A, lda, 32);
  v16h a2 = load_A_lds(A, lda, 64);
  v16h a3 = load_A_lds(A, lda, 96);
  v16h a4 = load_A_lds(Ap, 32, 0);
  v16h b0 = load_B_frag(fb);
  v16h b1 = load_B_frag(fb + step);
  v16h b2 = load_B_frag(fb + 2 * step);
  v16h b3 = load_B_frag(fb + 3 * step);
  v16h b4 = load_B_frag(fb + 4 * step);
  c = wmma16(a0, b0, c);
  c = wmma16(a1, b1, c);
  c = wmma16(a2, b2, c);
  c = wmma16(a3, b3, c);
  c = wmma16(a4, b4, c);
  return c;
}

// score-style GEMM: A row-major, B = transpose of row-major K-matrix (both LDS)
__device__ inline v8f gemm4_T(const _Float16* A, const _Float16* Krows, int n0) {
  v16h a0 = load_A_lds(A, 136, 0);
  v16h a1 = load_A_lds(A, 136, 32);
  v16h a2 = load_A_lds(A, 136, 64);
  v16h a3 = load_A_lds(A, 136, 96);
  v16h b0 = load_B_ldsT(Krows, 136, 0, n0);
  v16h b1 = load_B_ldsT(Krows, 136, 32, n0);
  v16h b2 = load_B_ldsT(Krows, 136, 64, n0);
  v16h b3 = load_B_ldsT(Krows, 136, 96, n0);
  v8f c = {};
  c = wmma16(a0, b0, c);
  c = wmma16(a1, b1, c);
  c = wmma16(a2, b2, c);
  c = wmma16(a3, b3, c);
  return c;
}

// ---------------------------------------------------------------------------
// Prep kernel: weights fp32 -> f16 pre-swizzled B fragments (512 halfs each).
//   Wq:[0,64) Wk:[64,128) Wq_o:[128,192) Wk_o:[192,256)
//   Wv(K->160):[256,296) Wv_o:[296,336) W1[:128]:[336,352) W1[128:]:[352,368)
// ---------------------------------------------------------------------------
__global__ __launch_bounds__(512) void prep_weights(
    const float* __restrict__ Wq, const float* __restrict__ Wk,
    const float* __restrict__ Wqo, const float* __restrict__ Wko,
    const float* __restrict__ Wv, const float* __restrict__ Wvo,
    const float* __restrict__ W1, _Float16* __restrict__ ws) {
  int f = blockIdx.x;
  const float* src; int Kreal, stride, ntiles, rowoff = 0, mbase;
  if (f < 64)        { src = Wq;  Kreal = 128; stride = 128; ntiles = 8; mbase = 0;   }
  else if (f < 128)  { src = Wk;  Kreal = 128; stride = 128; ntiles = 8; mbase = 64;  }
  else if (f < 192)  { src = Wqo; Kreal = 128; stride = 128; ntiles = 8; mbase = 128; }
  else if (f < 256)  { src = Wko; Kreal = 128; stride = 128; ntiles = 8; mbase = 192; }
  else if (f < 296)  { src = Wv;  Kreal = 136; stride = 128; ntiles = 8; mbase = 256; }
  else if (f < 336)  { src = Wvo; Kreal = 136; stride = 128; ntiles = 8; mbase = 296; }
  else if (f < 352)  { src = W1;  Kreal = 128; stride = 64;  ntiles = 4; mbase = 336; }
  else               { src = W1;  Kreal = 128; stride = 64;  ntiles = 4; mbase = 352; rowoff = 128; }
  int loc = f - mbase;
  int kt = loc / ntiles, nt = loc % ntiles;
  int l = threadIdx.x >> 4, e = threadIdx.x & 15;
  int K = kt * 32 + 16 * (l >> 4) + e;
  int n = nt * 16 + (l & 15);
  float v = (K < Kreal) ? src[(K + rowoff) * stride + n] : 0.0f;
  ws[(size_t)f * 512 + threadIdx.x] = (_Float16)v;
}

// ---------------------------------------------------------------------------
// Main kernel: one workgroup (8 wave32) per batch element.
// ---------------------------------------------------------------------------
__global__ __launch_bounds__(256) void critic_main(
    const float* __restrict__ states, const float* __restrict__ policies,
    const float* __restrict__ actions, const float* __restrict__ states_other,
    const float* __restrict__ actions_other, const float* __restrict__ W2,
    const _Float16* __restrict__ wf, float* __restrict__ out) {

  __shared__ __align__(32) unsigned char smem[56320];
  // Region A (P0-P1 inputs), later overlaid:
  _Float16* sS    = (_Float16*)(smem + 0);      // [16][136]
  _Float16* sAct  = (_Float16*)(smem + 4352);   // [16][32]
  _Float16* sPol  = (_Float16*)(smem + 5376);   // [16][32]
  _Float16* sSO   = (_Float16*)(smem + 6400);   // [32][136]
  _Float16* sActO = (_Float16*)(smem + 15104);  // [32][32]
  float*    sScore  = (float*)(smem + 0);       // [16][20]   (P2+)
  float*    sScoreO = (float*)(smem + 1280);    // [16][36]
  float*    sUZ     = (float*)(smem + 4096);    // [16][72]   (P4+)
  float*    sV      = (float*)(smem + 8704);    // [16][72]
  float*    sW2f    = (float*)(smem + 13312);   // [64]
  // Region B (P1-P2), later overlaid:
  _Float16* sQ    = (_Float16*)(smem + 17152);  // [16][136]
  _Float16* sK    = (_Float16*)(smem + 21504);  // [16][136]
  _Float16* sQO   = (_Float16*)(smem + 25856);  // [16][136]
  _Float16* sKO   = (_Float16*)(smem + 30208);  // [32][136]
  _Float16* sBase = (_Float16*)(smem + 17152);  // [16][136]  (P4a+)
  _Float16* sWavO = (_Float16*)(smem + 21504);  // [16][136]
  float*    sWf   = (float*)(smem + 25856);     // [16][16]   (P3+)
  _Float16* sW16  = (_Float16*)(smem + 26880);  // [16][40]
  _Float16* sWO16 = (_Float16*)(smem + 28160);  // [16][40]
  // Region C:
  _Float16* sAVA  = (_Float16*)(smem + 38912);  // [16][136]
  _Float16* sAVP  = (_Float16*)(smem + 43264);  // [16][136] -> diff
  _Float16* sAVO  = (_Float16*)(smem + 47616);  // [32][136]

  const int tid  = threadIdx.x;
  const int wave = tid >> 5;
  const int b    = blockIdx.x;

  float* out_val = out;
  float* out_w   = out + 128 * 256;
  float* out_wo  = out + 128 * 256 * 2;

  // -------- P0: stage activations as f16 --------
  {
    const float* Sg = states + (size_t)b * 2048;
    for (int e = tid; e < 2048; e += 256) sS[(e >> 7) * 136 + (e & 127)] = (_Float16)Sg[e];
    const float* Ag = actions + (size_t)b * 128;
    const float* Pg = policies + (size_t)b * 128;
    for (int e = tid; e < 512; e += 256) {
      int r = e >> 5, c = e & 31;
      sAct[r * 32 + c] = (_Float16)((c < 8) ? Ag[r * 8 + c] : 0.0f);
      sPol[r * 32 + c] = (_Float16)((c < 8) ? Pg[r * 8 + c] : 0.0f);
    }
    const float* SOg = states_other + (size_t)b * 4096;
    for (int e = tid; e < 4096; e += 256) sSO[(e >> 7) * 136 + (e & 127)] = (_Float16)SOg[e];
    const float* AOg = actions_other + (size_t)b * 256;
    for (int e = tid; e < 1024; e += 256) {
      int r = e >> 5, c = e & 31;
      sActO[r * 32 + c] = (_Float16)((c < 8) ? AOg[r * 8 + c] : 0.0f);
    }
  }
  __syncthreads();

  // -------- P1: projection + value GEMMs (72 tiles over 8 waves) --------
  for (int t = wave; t < 72; t += 8) {
    int nt = t & 7, n0 = nt * 16;
    if (t < 24) {                       // q, k, q_o = states @ {Wq,Wk,Wq_o}
      int which = t >> 3;
      _Float16* D = (which == 0) ? sQ : (which == 1) ? sK : sQO;
      v8f c = {};
      c = gemm4_frag(sS, 136, wf + (long)(which * 64 + nt) * 512, 4096, c);
      store_D_f16<false>(c, D, 136, n0);
    } else if (t < 40) {                // k_o = states_other @ Wk_o
      int mt = (t - 24) >> 3;
      v8f c = {};
      c = gemm4_frag(sSO + mt * 16 * 136, 136, wf + (long)(192 + nt) * 512, 4096, c);
      store_D_f16<false>(c, sKO + mt * 16 * 136, 136, n0);
    } else if (t < 56) {                // av_act / av_pol = tanh([S|act] @ Wv)
      int which = (t - 40) >> 3;
      v8f c = {};
      c = gemm5_frag(sS, 136, which ? sPol : sAct, wf + (long)(256 + nt) * 512, 4096, c);
      store_D_f16<true>(c, which ? sAVP : sAVA, 136, n0);
    } else {                            // av_o = tanh([SO|actO] @ Wv_o)
      int mt = (t - 56) >> 3;
      v8f c = {};
      c = gemm5_frag(sSO + mt * 16 * 136, 136, sActO + mt * 16 * 32,
                     wf + (long)(296 + nt) * 512, 4096, c);
      store_D_f16<true>(c, sAVO + mt * 16 * 136, 136, n0);
    }
  }
  __syncthreads();

  // -------- P2: attention scores (+ diff & W2 staging on idle waves) --------
  if (wave == 0) {                       // score = q @ k^T
    store_D_f32(gemm4_T(sQ, sK, 0), sScore, 20, 0);
  } else if (wave <= 2) {                // score_o = q_o @ k_o^T
    int nt = wave - 1;
    store_D_f32(gemm4_T(sQO, sKO, nt * 16), sScoreO, 36, nt * 16);
  } else {
    int idx = tid - 96;                  // 160 threads
    if (idx < 64) sW2f[idx] = W2[idx];
    for (int e = idx; e < 2048; e += 160) {   // diff = av_pol - av_act (in place)
      int r = e >> 7, c = e & 127;
      float d = (float)sAVP[r * 136 + c] - (float)sAVA[r * 136 + c];
      sAVP[r * 136 + c] = (_Float16)d;
    }
  }
  __syncthreads();

  // -------- P3: softmaxes, write weight outputs --------
  {
    const float scale = 0.08838834764831845f;  // 1/sqrt(128)
    if (tid < 16) {                      // weight: softmax over keys (rows)
      int i = tid;
      float x[16], m = -1e30f;
      for (int j = 0; j < 16; ++j) { x[j] = sScore[i * 20 + j] * scale; m = fmaxf(m, x[j]); }
      float s = 0.f;
      for (int j = 0; j < 16; ++j) { x[j] = __expf(x[j] - m); s += x[j]; }
      float inv = __fdividef(1.f, s);
      for (int j = 0; j < 16; ++j) {
        float w = x[j] * inv;
        sWf[i * 16 + j] = w;
        sW16[i * 40 + j] = (_Float16)w;
        out_w[(size_t)b * 256 + i * 16 + j] = w;
      }
      for (int j = 16; j < 32; ++j) sW16[i * 40 + j] = (_Float16)0.0f;  // K pad
    } else if (tid < 48) {               // weight_o: softmax over agent axis (cols)
      int mcol = tid - 16;
      float x[16], m = -1e30f;
      for (int i = 0; i < 16; ++i) { x[i] = sScoreO[i * 36 + mcol] * scale; m = fmaxf(m, x[i]); }
      float s = 0.f;
      for (int i = 0; i < 16; ++i) { x[i] = __expf(x[i] - m); s += x[i]; }
      float inv = __fdividef(1.f, s);
      for (int i = 0; i < 16; ++i) {
        float w = x[i] * inv;
        sWO16[i * 40 + mcol] = (_Float16)w;
        out_wo[(size_t)b * 512 + i * 32 + mcol] = w;
      }
    }
  }
  __syncthreads();

  // -------- P4a: base = weight @ av_act ; wav_o = weight_o @ av_o --------
  for (int t = wave; t < 16; t += 8) {
    bool isBase = (t < 8);
    int n0 = (t & 7) * 16;
    v16h a = load_A_lds(isBase ? sW16 : sWO16, 40, 0);
    v16h bb = load_B_ldscol(isBase ? sAVA : sAVO, 136, 0, n0, isBase ? 16 : 32);
    v8f c = {};
    c = wmma16(a, bb, c);
    store_D_f16<false>(c, isBase ? sBase : sWavO, 136, n0);
  }
  __syncthreads();

  // -------- P4b: UZ = base@W1t + wav_o@W1b ;  V = diff@W1t --------
  if (wave < 4) {
    int nt = wave;
    v8f c = {};
    c = gemm4_frag(sBase, 136, wf + (long)(336 * 512) + nt * 512, 2048, c);
    c = gemm4_frag(sWavO, 136, wf + (long)(352 * 512) + nt * 512, 2048, c);
    store_D_f32(c, sUZ, 72, nt * 16);
  } else {
    int nt = wave - 4;
    v8f c = {};
    c = gemm4_frag(sAVP, 136, wf + (long)(336 * 512) + nt * 512, 2048, c);
    store_D_f32(c, sV, 72, nt * 16);
  }
  __syncthreads();

  // -------- P5: value[i,j] = W2 . leaky_relu(UZ[i] + w[i,j]*V[j]) --------
  {
    int i = tid >> 4, j = tid & 15;
    float wij = sWf[i * 16 + j];
    float acc = 0.f;
#pragma unroll 8
    for (int d = 0; d < 64; ++d) {
      float x = sUZ[i * 72 + d] + wij * sV[j * 72 + d];
      x = fmaxf(x, 0.01f * x);            // leaky_relu slope 0.01
      acc += x * sW2f[d];
    }
    out_val[(size_t)b * 256 + i * 16 + j] = acc;
  }
}

extern "C" void kernel_launch(void* const* d_in, const int* in_sizes, int n_in,
                              void* d_out, int out_size, void* d_ws, size_t ws_size,
                              hipStream_t stream) {
  const float* states        = (const float*)d_in[0];
  const float* policies      = (const float*)d_in[1];
  const float* actions       = (const float*)d_in[2];
  const float* states_other  = (const float*)d_in[3];
  const float* actions_other = (const float*)d_in[4];
  const float* Wk  = (const float*)d_in[5];
  const float* Wq  = (const float*)d_in[6];
  const float* Wv  = (const float*)d_in[7];
  const float* Wko = (const float*)d_in[8];
  const float* Wqo = (const float*)d_in[9];
  const float* Wvo = (const float*)d_in[10];
  const float* W1  = (const float*)d_in[11];
  const float* W2  = (const float*)d_in[12];
  _Float16* wsf16 = (_Float16*)d_ws;   // needs 368*512*2 = 376,832 bytes

  prep_weights<<<368, 512, 0, stream>>>(Wq, Wk, Wqo, Wko, Wv, Wvo, W1, wsf16);
  critic_main<<<128, 256, 0, stream>>>(states, policies, actions, states_other,
                                       actions_other, W2, wsf16, (float*)d_out);
}
